// GATEncoder_89481348645582
// MI455X (gfx1250) — matrix operand order repaired and checked by reading, not compiled, not run on hardware
//
#include <hip/hip_runtime.h>

// ---------------------------------------------------------------------------
// GAT encoder (3 layers) for MI455X / gfx1250.
//   GEMMs       : v_wmma_f32_16x16x32_bf16, one 16x16 C tile per wave32.
//   Edge softmax: atomic-max (order-preserving uint encode) + atomic-add.
//   Aggregation : wave per edge (all 4 heads), grid-stride + global_prefetch
//                 of the next gather row, float2 loads, fp32 atomic scatter.
// ---------------------------------------------------------------------------

#define N_NODES 50000
#define N_EDGES 800000
#define HEADS   4
#define DH      64
#define F       256          // HEADS * DH
#define NEG_SLOPE 0.2f

typedef __attribute__((ext_vector_type(16))) __bf16 v16bf;
typedef __attribute__((ext_vector_type(8)))  float  v8f;

// ---------------------------------------------------------------- helpers ---
__device__ __forceinline__ unsigned short f32_to_bf16(float f) {
    unsigned u = __float_as_uint(f);
    u += 0x7fffu + ((u >> 16) & 1u);          // round-to-nearest-even
    return (unsigned short)(u >> 16);
}
__device__ __forceinline__ unsigned flipf(float f) {   // order-preserving
    unsigned u = __float_as_uint(f);
    return (u & 0x80000000u) ? ~u : (u | 0x80000000u);
}
__device__ __forceinline__ float unflipf(unsigned u) {
    return (u & 0x80000000u) ? __uint_as_float(u & 0x7fffffffu)
                             : __uint_as_float(~u);
}

// ------------------------------------------------------------------ fills ---
__global__ void fill_u32_kernel(unsigned* __restrict__ p, unsigned v, int count) {
    int i = blockIdx.x * blockDim.x + threadIdx.x;
    if (i < count) p[i] = v;
}

// -------------------------------------------------------------- converts ---
__global__ void cvt_bf16_kernel(const float* __restrict__ src,
                                unsigned short* __restrict__ dst, int count) {
    int i = blockIdx.x * blockDim.x + threadIdx.x;
    if (i < count) dst[i] = f32_to_bf16(src[i]);
}

// W is [K, 256] row-major (fp32) -> Wt is [256, K] row-major (bf16)
__global__ void cvt_wt_kernel(const float* __restrict__ W,
                              unsigned short* __restrict__ Wt, int K) {
    int i = blockIdx.x * blockDim.x + threadIdx.x;
    if (i >= 256 * K) return;
    int n = i / K, k = i - n * K;
    Wt[i] = f32_to_bf16(W[(size_t)k * 256 + n]);
}

// ---------------------------------------------------------- WMMA bf16 GEMM --
// C[M,256] = A[M,K] (bf16, row-major) * Wt[256,K]^T (bf16, row-major Wt)
// One 16x16 output tile per wave; 4 waves / block; M % 16 == 0, K % 32 == 0.
union BfPack { uint4 u; __bf16 h[8]; };

__global__ __launch_bounds__(128)
void gemm_bf16_wmma_kernel(const unsigned short* __restrict__ Abf,
                           const unsigned short* __restrict__ Wt,
                           float* __restrict__ C, int M, int K) {
    const int lane = threadIdx.x & 31;
    const int wave = threadIdx.x >> 5;
    const int tile = blockIdx.x * 4 + wave;        // wave-uniform
    const int tm = tile >> 4;                       // 16 n-tiles (256/16)
    const int tn = tile & 15;
    if (tm * 16 >= M) return;                       // wave-uniform branch

    const int half = lane >> 4;                     // 0 or 1
    const int l16  = lane & 15;

    const unsigned short* arow = Abf + (size_t)(tm * 16 + l16) * K;
    const unsigned short* brow = Wt  + (size_t)(tn * 16 + l16) * K;

    v8f acc = {};
    for (int k0 = 0; k0 < K; k0 += 32) {
        BfPack a0, a1, b0, b1;
        // A 16x32 layout: VGPR0-3 -> K = k0 + half*8 + [0..7]
        //                 VGPR4-7 -> K = k0 + 16 + half*8 + [0..7]
        a0.u = *(const uint4*)(arow + k0 + half * 8);
        a1.u = *(const uint4*)(arow + k0 + 16 + half * 8);
        // B 32x16 layout: VGPR0-3 -> K = k0 + half*8 + [0..7], VGPR4-7 -> K = k0 + 16 + half*8 + [0..7] at col l16
        b0.u = *(const uint4*)(brow + k0 + half * 8);
        b1.u = *(const uint4*)(brow + k0 + 16 + half * 8);
        v16bf av, bv;
#pragma unroll
        for (int j = 0; j < 8; ++j) {
            av[j]     = a0.h[j];
            av[8 + j] = a1.h[j];
            bv[j]     = b0.h[j];
            bv[8 + j] = b1.h[j];
        }
        acc = __builtin_amdgcn_wmma_f32_16x16x32_bf16(
                  false, av, false, bv, (short)0, acc, false, false);
    }
    // C 16x16 f32 layout: VGPR r -> row half*8 + r, col l16
    float* crow = C + (size_t)(tm * 16 + half * 8) * 256 + tn * 16 + l16;
#pragma unroll
    for (int r = 0; r < 8; ++r) crow[(size_t)r * 256] = acc[r];
}

// ---------------------------------------------------------------- el / er ---
__global__ void el_er_kernel(const float* __restrict__ h,
                             const float* __restrict__ al,
                             const float* __restrict__ ar,
                             float* __restrict__ el, float* __restrict__ er) {
    int i = blockIdx.x * blockDim.x + threadIdx.x;     // node*4 + head
    if (i >= N_NODES * HEADS) return;
    int head = i & 3, node = i >> 2;
    const float4* hv  = (const float4*)(h  + (size_t)node * F + head * DH);
    const float4* alv = (const float4*)(al + head * DH);
    const float4* arv = (const float4*)(ar + head * DH);
    float sl = 0.f, sr = 0.f;
#pragma unroll
    for (int j = 0; j < DH / 4; ++j) {
        float4 x = hv[j], a = alv[j], b = arv[j];
        sl += x.x * a.x + x.y * a.y + x.z * a.z + x.w * a.w;
        sr += x.x * b.x + x.y * b.y + x.z * b.z + x.w * b.w;
    }
    el[i] = sl; er[i] = sr;
}

// ----------------------------------------------------- edge segment-max -----
__global__ void edge_max_kernel(const float* __restrict__ el,
                                const float* __restrict__ er,
                                const int* __restrict__ src,
                                const int* __restrict__ dst,
                                unsigned* __restrict__ m_u) {
    int e = blockIdx.x * blockDim.x + threadIdx.x;
    if (e >= N_EDGES) return;
    int s = src[e], d = dst[e];
    float4 a = *(const float4*)(el + (size_t)s * 4);
    float4 b = *(const float4*)(er + (size_t)d * 4);
    float ev[4] = {a.x + b.x, a.y + b.y, a.z + b.z, a.w + b.w};
#pragma unroll
    for (int h = 0; h < 4; ++h) {
        float t = ev[h] > 0.f ? ev[h] : NEG_SLOPE * ev[h];
        atomicMax(&m_u[(size_t)d * 4 + h], flipf(t));
    }
}

// ------------------------------------------------- edge exp + segment-sum ---
__global__ void edge_expsum_kernel(const float* __restrict__ el,
                                   const float* __restrict__ er,
                                   const int* __restrict__ src,
                                   const int* __restrict__ dst,
                                   const unsigned* __restrict__ m_u,
                                   float* __restrict__ ssum,
                                   float* __restrict__ ee) {
    int e = blockIdx.x * blockDim.x + threadIdx.x;
    if (e >= N_EDGES) return;
    int s = src[e], d = dst[e];
    float4 a = *(const float4*)(el + (size_t)s * 4);
    float4 b = *(const float4*)(er + (size_t)d * 4);
    float ev[4] = {a.x + b.x, a.y + b.y, a.z + b.z, a.w + b.w};
#pragma unroll
    for (int h = 0; h < 4; ++h) {
        float t = ev[h] > 0.f ? ev[h] : NEG_SLOPE * ev[h];
        float m = unflipf(m_u[(size_t)d * 4 + h]);
        float x = __expf(t - m);
        ee[(size_t)e * 4 + h] = x;
        atomicAdd(&ssum[(size_t)d * 4 + h], x);
    }
}

// --------------------------------------- message aggregation (scatter-add) --
// One wave32 per edge handles all 4 heads (64 floats each -> float2 per lane).
// Grid-stride loop; prefetch next edge's gather row (global_prefetch_b8).
#define AGG_BLOCKS 4096
__global__ __launch_bounds__(256)
void edge_agg_kernel(const float* __restrict__ ee,
                     const float* __restrict__ ssum,
                     const int* __restrict__ src,
                     const int* __restrict__ dst,
                     const float* __restrict__ h,
                     float* __restrict__ accum) {
    const int lane   = threadIdx.x & 31;
    const int stride = AGG_BLOCKS * 8;                       // total waves
    int e = (blockIdx.x * 256 + threadIdx.x) >> 5;
    for (; e < N_EDGES; e += stride) {
        int s = src[e], d = dst[e];

        // prefetch the next iteration's gather row while we do this one
        int en = e + stride;
        if (en < N_EDGES) {
            const float* hn = h + (size_t)src[en] * F;
            __builtin_prefetch(hn + lane * 8, 0, 0);         // 32B/lane covers 1KB row
        }

        float4 eev = *(const float4*)(ee   + (size_t)e * 4);
        float4 sv  = *(const float4*)(ssum + (size_t)d * 4);
        float alpha[4] = {eev.x / sv.x, eev.y / sv.y, eev.z / sv.z, eev.w / sv.w};

        const float* hrow = h     + (size_t)s * F;
        float*       orow = accum + (size_t)d * F;
#pragma unroll
        for (int hh = 0; hh < 4; ++hh) {
            float2 hv = *(const float2*)(hrow + hh * DH + lane * 2);
            float* op = orow + hh * DH + lane * 2;
            atomicAdd(op,     alpha[hh] * hv.x);
            atomicAdd(op + 1, alpha[hh] * hv.y);
        }
    }
}

// --------------------------------------------- epilogues -------------------
__global__ void post_elu_kernel(const float* __restrict__ accum,
                                const float* __restrict__ bias,
                                float* __restrict__ feat,
                                unsigned short* __restrict__ featb) {
    int i = blockIdx.x * blockDim.x + threadIdx.x;
    if (i >= N_NODES * F) return;
    float v = accum[i] + bias[i & (F - 1)];
    float f = v > 0.f ? v : (__expf(v) - 1.f);      // ELU(alpha=1)
    feat[i]  = f;
    featb[i] = f32_to_bf16(f);
}

__global__ void post_final_kernel(const float* __restrict__ accum,
                                  const float* __restrict__ bias,
                                  const float* __restrict__ feat,   // residual
                                  float* __restrict__ out) {
    int i = blockIdx.x * blockDim.x + threadIdx.x;
    if (i >= N_NODES * DH) return;
    int n = i >> 6, d = i & 63;
    float s = 0.f;
#pragma unroll
    for (int hh = 0; hh < 4; ++hh) {
        int idx = n * F + hh * DH + d;
        s += accum[idx] + bias[hh * DH + d] + feat[idx];
    }
    out[i] = 0.25f * s;
}

// ---------------------------------------------------------------------------
extern "C" void kernel_launch(void* const* d_in, const int* in_sizes, int n_in,
                              void* d_out, int out_size, void* d_ws, size_t ws_size,
                              hipStream_t stream) {
    (void)in_sizes; (void)n_in; (void)out_size; (void)ws_size;

    const float* x   = (const float*)d_in[0];
    const float* Wl[3]  = {(const float*)d_in[1], (const float*)d_in[5], (const float*)d_in[9]};
    const float* all[3] = {(const float*)d_in[2], (const float*)d_in[6], (const float*)d_in[10]};
    const float* arl[3] = {(const float*)d_in[3], (const float*)d_in[7], (const float*)d_in[11]};
    const float* bl[3]  = {(const float*)d_in[4], (const float*)d_in[8], (const float*)d_in[12]};
    const int* src = (const int*)d_in[13];
    const int* dst = (const int*)d_in[14];
    float* out = (float*)d_out;

    // -------- workspace carve-out (all offsets 256B aligned) --------
    char* p = (char*)d_ws;
    auto alloc = [&](size_t bytes) -> char* {
        char* r = p; p += (bytes + 255) & ~(size_t)255; return r;
    };
    float*          feat   = (float*)         alloc((size_t)N_NODES * F * 4); // fp32 features (residual)
    unsigned short* featb  = (unsigned short*)alloc((size_t)N_NODES * F * 2); // bf16 GEMM A
    float*          hbuf   = (float*)         alloc((size_t)N_NODES * F * 4); // h = A@W
    float*          accum  = (float*)         alloc((size_t)N_NODES * F * 4); // segment-sum
    unsigned short* Wt     = (unsigned short*)alloc((size_t)256 * 256 * 2);   // bf16 W^T
    float*          el     = (float*)         alloc((size_t)N_NODES * 4 * 4);
    float*          er     = (float*)         alloc((size_t)N_NODES * 4 * 4);
    unsigned*       m_u    = (unsigned*)      alloc((size_t)N_NODES * 4 * 4);
    float*          ssum   = (float*)         alloc((size_t)N_NODES * 4 * 4);
    float*          ee     = (float*)         alloc((size_t)N_EDGES * 4 * 4);

    const int BLK = 256;
    auto nb = [](int count, int blk) { return (count + blk - 1) / blk; };

    // layer-0 input: convert x (N x 64) to bf16
    cvt_bf16_kernel<<<nb(N_NODES * DH, BLK), BLK, 0, stream>>>(x, featb, N_NODES * DH);

    for (int l = 0; l < 3; ++l) {
        const int K = (l == 0) ? DH : F;

        // weights -> bf16 transposed
        cvt_wt_kernel<<<nb(256 * K, BLK), BLK, 0, stream>>>(Wl[l], Wt, K);

        // h = A @ W  (WMMA bf16)
        const int tiles = (N_NODES / 16) * 16;           // 3125 m-tiles * 16 n-tiles
        gemm_bf16_wmma_kernel<<<tiles / 4, 128, 0, stream>>>(featb, Wt, hbuf, N_NODES, K);

        // attention logits per node
        el_er_kernel<<<nb(N_NODES * HEADS, BLK), BLK, 0, stream>>>(hbuf, all[l], arl[l], el, er);

        // reset segment state + accumulator
        fill_u32_kernel<<<nb(N_NODES * HEADS, BLK), BLK, 0, stream>>>(m_u, 0u, N_NODES * HEADS);
        fill_u32_kernel<<<nb(N_NODES * HEADS, BLK), BLK, 0, stream>>>((unsigned*)ssum, 0u, N_NODES * HEADS);
        fill_u32_kernel<<<nb(N_NODES * F, BLK), BLK, 0, stream>>>((unsigned*)accum, 0u, N_NODES * F);

        // segment softmax over dst
        edge_max_kernel<<<nb(N_EDGES, BLK), BLK, 0, stream>>>(el, er, src, dst, m_u);
        edge_expsum_kernel<<<nb(N_EDGES, BLK), BLK, 0, stream>>>(el, er, src, dst, m_u, ssum, ee);

        // weighted scatter-add of messages (persistent waves + prefetch)
        edge_agg_kernel<<<AGG_BLOCKS, 256, 0, stream>>>(ee, ssum, src, dst, hbuf, accum);

        if (l < 2) {
            // bias + ELU, write fp32 features (residual source) + bf16 GEMM input
            post_elu_kernel<<<nb(N_NODES * F, BLK), BLK, 0, stream>>>(accum, bl[l], feat, featb);
        } else {
            // bias + residual + head-mean -> final [N, 64]
            post_final_kernel<<<nb(N_NODES * DH, BLK), BLK, 0, stream>>>(accum, bl[l], feat, out);
        }
    }
}
